// LinearizedAttention_15006615733271
// MI455X (gfx1250) — matrix operand order repaired and checked by reading, not compiled, not run on hardware
//
#include <hip/hip_runtime.h>

// ---------------------------------------------------------------------------
// Linearized attention for MI455X (gfx1250, wave32, WMMA + async-LDS,
// double-buffered).
//
// Shapes: B=2, H=8, N=2048, dk=64, D=512.
// d_out = [ out : B*N*512 fp32 ][ attention_weights : B*H*N*N fp32 ]
//
// Workspace layout (bytes):
//   Qf   fp32 [B,H,N,64]   off 0        (8 MiB)
//   Kf   fp32 [B,H,N,64]   off 8  MiB   (8 MiB)
//   Vt   bf16 [B,H,64,N]   off 16 MiB   (4 MiB)   (pre-transposed V)
//   Qn   bf16 [B,H,N,64]   off 20 MiB   (4 MiB)
//   Kn   bf16 [B,H,N,64]   off 24 MiB   (4 MiB)
//   Yctx fp32 [B,N,512]    off 28 MiB   (8 MiB)
//   Ksum fp32 [B*H,64]     off 36 MiB   (4 KiB)
// ---------------------------------------------------------------------------

typedef unsigned short u16;
typedef unsigned int   u32;
typedef __attribute__((ext_vector_type(16))) __bf16 v16bf;
typedef __attribute__((ext_vector_type(8)))  float  v8f;
typedef __attribute__((ext_vector_type(2)))  float  f32x2;
typedef __attribute__((ext_vector_type(2)))  __bf16 bf16x2;

union Frag { uint4 q[2]; v16bf v; };
static_assert(sizeof(Frag) == 32, "frag size");

// Native bf16 converts (v_cvt_pk_bf16_f32 on gfx1250), RNE.
__device__ __forceinline__ u16 f2bf(float f) {
  __bf16 b = (__bf16)f;
  return __builtin_bit_cast(u16, b);
}
__device__ __forceinline__ u32 pk2(float a, float b) {
  f32x2 x; x[0] = a; x[1] = b;
  bf16x2 r = __builtin_convertvector(x, bf16x2);
  return __builtin_bit_cast(u32, r);
}
__device__ __forceinline__ v8f wmma_bf16(const v16bf& a, const v16bf& b, const v8f& c) {
  return __builtin_amdgcn_wmma_f32_16x16x32_bf16(false, a, false, b, (short)0, c,
                                                 false, false);
}
// Async global->LDS copy of 16 bytes (ASYNCcnt tracked).
__device__ __forceinline__ void async_copy16(u32 lds_off, const void* gaddr) {
  asm volatile("global_load_async_to_lds_b128 %0, %1, off"
               :: "v"(lds_off), "v"(gaddr) : "memory");
}
__device__ __forceinline__ u32 lds_off(const void* p) {
  return (u32)(uintptr_t)p;   // LDS aperture: addr[31:0] is the LDS offset
}

// ---------------------------------------------------------------------------
// GEMM: Y = X[4096x512] @ W^T[512x512] + bias, with per-MODE epilogue.
// MODE 0/1: Q/K -> elu+1, fp32 [B,H,N,64];  MODE 2: V -> bf16 [B,H,64,N];
// MODE 3: output projection -> fp32 [4096,512].
// Block 256 threads (8 waves). Tile 128(M) x 64(N), K-step 32.
// ---------------------------------------------------------------------------
template <int MODE>
__global__ __launch_bounds__(256) void gemm_proj(const float* __restrict__ X,
                                                 const float* __restrict__ W,
                                                 const float* __restrict__ bias,
                                                 void* __restrict__ Yout) {
  __shared__ __align__(16) u16 As[128 * 32];   // [m][k]  8 KiB
  __shared__ __align__(16) u16 Bs[64 * 32];    // [n][k]  4 KiB

  const int tid  = threadIdx.x;
  const int wave = tid >> 5;
  const int lane = tid & 31;
  const int nl   = lane & 15;
  const int hi   = lane >> 4;
  const int m0   = blockIdx.x * 128;
  const int o0   = blockIdx.y * 64;

  const int arow = tid >> 1;            // A staging: 128 rows x 2 k-halves
  const int akh  = (tid & 1) * 16;
  const int brow = tid >> 2;            // B staging: 64 rows x 4 k-quarters
  const int bkq  = (tid & 3) * 8;

  v8f acc0 = {}, acc1 = {}, acc2 = {}, acc3 = {};

  for (int k0 = 0; k0 < 512; k0 += 32) {
    // speculative prefetch of next K tile (safe: dropped on fault)
    __builtin_prefetch(X + (size_t)(m0 + arow) * 512 + k0 + 32 + akh, 0, 0);
    __builtin_prefetch(W + (size_t)(o0 + brow) * 512 + k0 + 32 + bkq, 0, 0);
    // ---- stage A tile (fp32 -> bf16, native cvt) ----
    {
      const float4* s = (const float4*)(X + (size_t)(m0 + arow) * 512 + k0 + akh);
      float4 f0 = s[0], f1 = s[1], f2 = s[2], f3 = s[3];
      uint4 p0, p1;
      p0.x = pk2(f0.x, f0.y); p0.y = pk2(f0.z, f0.w);
      p0.z = pk2(f1.x, f1.y); p0.w = pk2(f1.z, f1.w);
      p1.x = pk2(f2.x, f2.y); p1.y = pk2(f2.z, f2.w);
      p1.z = pk2(f3.x, f3.y); p1.w = pk2(f3.z, f3.w);
      *(uint4*)&As[arow * 32 + akh]     = p0;
      *(uint4*)&As[arow * 32 + akh + 8] = p1;
    }
    // ---- stage B tile: B(k,n) = W[o0+n][k] ----
    {
      const float4* s = (const float4*)(W + (size_t)(o0 + brow) * 512 + k0 + bkq);
      float4 f0 = s[0], f1 = s[1];
      uint4 p;
      p.x = pk2(f0.x, f0.y); p.y = pk2(f0.z, f0.w);
      p.z = pk2(f1.x, f1.y); p.w = pk2(f1.z, f1.w);
      *(uint4*)&Bs[brow * 32 + bkq] = p;
    }
    __syncthreads();

    // ---- fragments: two contiguous b128 LDS loads each ----
    Frag fa;
    {
      const int ar = wave * 16 + nl;
      const int ko = hi * 8;
      fa.q[0] = *(const uint4*)&As[ar * 32 + ko];
      fa.q[1] = *(const uint4*)&As[ar * 32 + 16 + ko];
    }
    const int ko2 = hi * 8;
    Frag fb0, fb1, fb2, fb3;
    fb0.q[0] = *(const uint4*)&Bs[(0  + nl) * 32 + ko2];
    fb0.q[1] = *(const uint4*)&Bs[(0  + nl) * 32 + ko2 + 16];
    fb1.q[0] = *(const uint4*)&Bs[(16 + nl) * 32 + ko2];
    fb1.q[1] = *(const uint4*)&Bs[(16 + nl) * 32 + ko2 + 16];
    fb2.q[0] = *(const uint4*)&Bs[(32 + nl) * 32 + ko2];
    fb2.q[1] = *(const uint4*)&Bs[(32 + nl) * 32 + ko2 + 16];
    fb3.q[0] = *(const uint4*)&Bs[(48 + nl) * 32 + ko2];
    fb3.q[1] = *(const uint4*)&Bs[(48 + nl) * 32 + ko2 + 16];

    acc0 = wmma_bf16(fa.v, fb0.v, acc0);
    acc1 = wmma_bf16(fa.v, fb1.v, acc1);
    acc2 = wmma_bf16(fa.v, fb2.v, acc2);
    acc3 = wmma_bf16(fa.v, fb3.v, acc3);
    __syncthreads();
  }

  // ---- epilogue ----
  const float bv0 = bias[o0 + 0  + nl];
  const float bv1 = bias[o0 + 16 + nl];
  const float bv2 = bias[o0 + 32 + nl];
  const float bv3 = bias[o0 + 48 + nl];
  const int h = blockIdx.y;

#pragma unroll
  for (int r = 0; r < 8; ++r) {
    const int m = m0 + wave * 16 + hi * 8 + r;   // global row in [0,4096)
    const int b = m >> 11;
    const int n = m & 2047;
    float v0 = acc0[r] + bv0, v1 = acc1[r] + bv1;
    float v2 = acc2[r] + bv2, v3 = acc3[r] + bv3;
    if (MODE <= 1) {  // elu(x)+1 feature map, store fp32 [B,H,N,64]
      v0 = v0 > 0.f ? v0 + 1.f : __expf(v0);
      v1 = v1 > 0.f ? v1 + 1.f : __expf(v1);
      v2 = v2 > 0.f ? v2 + 1.f : __expf(v2);
      v3 = v3 > 0.f ? v3 + 1.f : __expf(v3);
      float* Y = (float*)Yout;
      const size_t base = ((size_t)(b * 8 + h) * 2048 + n) * 64;
      Y[base + 0  + nl] = v0;
      Y[base + 16 + nl] = v1;
      Y[base + 32 + nl] = v2;
      Y[base + 48 + nl] = v3;
    } else if (MODE == 2) {  // V: bf16 transposed [B,H,64,N]
      u16* Y = (u16*)Yout;
      const size_t hb = (size_t)(b * 8 + h) * 64;
      Y[(hb + 0  + nl) * 2048 + n] = f2bf(v0);
      Y[(hb + 16 + nl) * 2048 + n] = f2bf(v1);
      Y[(hb + 32 + nl) * 2048 + n] = f2bf(v2);
      Y[(hb + 48 + nl) * 2048 + n] = f2bf(v3);
    } else {  // output projection: fp32 [4096,512]
      float* Y = (float*)Yout;
      const size_t base = (size_t)m * 512 + o0;
      Y[base + 0  + nl] = v0;
      Y[base + 16 + nl] = v1;
      Y[base + 32 + nl] = v2;
      Y[base + 48 + nl] = v3;
    }
  }
}

// ---------------------------------------------------------------------------
// Q normalization: Qn = Qf / (row-sum over 64 features + eps), bf16 out.
// ---------------------------------------------------------------------------
__global__ __launch_bounds__(256) void rownorm_q(const float* __restrict__ Qf,
                                                 u16* __restrict__ Qn) {
  const int row  = blockIdx.x * 8 + (threadIdx.x >> 5);
  const int lane = threadIdx.x & 31;
  const float2 v = *(const float2*)(Qf + (size_t)row * 64 + lane * 2);
  float s = v.x + v.y;
#pragma unroll
  for (int m = 16; m >= 1; m >>= 1) s += __shfl_xor(s, m);
  const float inv = 1.0f / (s + 1e-8f);
  *(u32*)(Qn + (size_t)row * 64 + lane * 2) = pk2(v.x * inv, v.y * inv);
}

// K column sums over sequence dim: Ksum[bh][d] = sum_n Kf[bh][n][d].
__global__ __launch_bounds__(256) void colsum_k(const float* __restrict__ Kf,
                                                float* __restrict__ Ksum) {
  __shared__ float red[256];
  const int bh    = blockIdx.x;
  const int d     = threadIdx.x & 63;
  const int chunk = threadIdx.x >> 6;
  const float* p  = Kf + (size_t)bh * 2048 * 64 + (size_t)chunk * 512 * 64 + d;
  float s = 0.f;
  for (int n = 0; n < 512; ++n) s += p[(size_t)n * 64];
  red[threadIdx.x] = s;
  __syncthreads();
  if (threadIdx.x < 64) {
    Ksum[bh * 64 + threadIdx.x] =
        red[threadIdx.x] + red[64 + threadIdx.x] + red[128 + threadIdx.x] +
        red[192 + threadIdx.x];
  }
}

// Kn = Kf / (Ksum + eps), bf16 out.
__global__ __launch_bounds__(256) void norm_k(const float* __restrict__ Kf,
                                              const float* __restrict__ Ksum,
                                              u16* __restrict__ Kn) {
  const size_t idx = ((size_t)blockIdx.x * 256 + threadIdx.x) * 2;
  const int bh = (int)(idx >> 17);
  const int d  = (int)(idx & 63);
  const float s0 = Ksum[bh * 64 + d];
  const float s1 = Ksum[bh * 64 + d + 1];
  const float v0 = Kf[idx]     / (s0 + 1e-8f);
  const float v1 = Kf[idx + 1] / (s1 + 1e-8f);
  *(u32*)(Kn + idx) = pk2(v0, v1);
}

// ---------------------------------------------------------------------------
// Fused attention:  A = Qn @ Kn^T  (nontemporal-streamed to d_out, fp32)
//                   Yctx += A @ V  (A re-fed via bf16 LDS staging)
// Kn/Vt tiles double-buffered in LDS via global_load_async_to_lds_b128:
// the fill for step i+1 is issued right after the barrier of step i, so the
// copy latency hides behind 8 WMMAs + NT stores. One barrier per step.
// Block = 128 threads (4 waves); each wave owns 16 rows; m-step 32.
// ---------------------------------------------------------------------------
__global__ __launch_bounds__(128) void attn_kernel(
    const u16* __restrict__ Qn, const u16* __restrict__ Kn,
    const u16* __restrict__ Vt, float* __restrict__ attnOut,
    float* __restrict__ Yctx) {
  __shared__ __align__(16) u16 Kbuf[2][32 * 64];  // [m_local][d]   2 x 4 KiB
  __shared__ __align__(16) u16 Vbuf[2][64 * 32];  // [d][m_local]   2 x 4 KiB
  __shared__ __align__(16) u16 Sbuf[4][16 * 32];  // per-wave A-tile 4 KiB

  const int tid  = threadIdx.x;
  const int wave = tid >> 5;
  const int lane = tid & 31;
  const int nl   = lane & 15;
  const int hi   = lane >> 4;
  const int bh   = blockIdx.y;
  const int rowbase = blockIdx.x * 64 + wave * 16;

  // Q fragments (K=64 -> two 16x16x32 A-fragments), loaded once.
  Frag faq0, faq1;
  {
    const u16* q = Qn + ((size_t)bh * 2048 + rowbase + nl) * 64;
    const int ko = hi * 8;
    faq0.q[0] = *(const uint4*)(q + 0  + ko);
    faq0.q[1] = *(const uint4*)(q + 16 + ko);
    faq1.q[0] = *(const uint4*)(q + 32 + ko);
    faq1.q[1] = *(const uint4*)(q + 48 + ko);
  }

  v8f o0 = {}, o1 = {}, o2 = {}, o3 = {};
  float* aout    = attnOut + (size_t)bh * 2048 * 2048;
  const u16* Kb  = Kn + (size_t)bh * 2048 * 64;
  const u16* Vb  = Vt + (size_t)bh * 64 * 2048;
  u16* sb        = &Sbuf[wave][0];
  const int ko   = hi * 8;
  const int ko2  = hi * 16;
  const u32 kbase0 = lds_off(&Kbuf[0][0]);
  const u32 kbase1 = lds_off(&Kbuf[1][0]);
  const u32 vbase0 = lds_off(&Vbuf[0][0]);
  const u32 vbase1 = lds_off(&Vbuf[1][0]);
  const int vrow  = tid >> 2;          // 0..31 (and +32 for second chunk)
  const int vcc   = (tid & 3) * 16;    // byte offset within 64B row

  // Cooperative async fill of Kn tile (contiguous 4KB) + Vt tile (64 rows).
  auto fill = [&](int c, int mm) {
    const u32 kb = c ? kbase1 : kbase0;
    const u32 vb = c ? vbase1 : vbase0;
    const char* gk = (const char*)(Kb + (size_t)mm * 64);
    async_copy16(kb + tid * 16,         gk + tid * 16);
    async_copy16(kb + (tid + 128) * 16, gk + (tid + 128) * 16);
    const char* gv0 = (const char*)(Vb + (size_t)vrow * 2048 + mm);
    const char* gv1 = (const char*)(Vb + (size_t)(vrow + 32) * 2048 + mm);
    async_copy16(vb + vrow * 64 + vcc,        gv0 + vcc);
    async_copy16(vb + (vrow + 32) * 64 + vcc, gv1 + vcc);
  };

  fill(0, 0);   // prologue

  for (int mm = 0; mm < 2048; mm += 32) {
    const int cur = (mm >> 5) & 1;
    asm volatile("s_wait_asynccnt 0x0" ::: "memory");  // fill(cur) landed
    __syncthreads();   // all waves' shares landed; prev compute done
    if (mm + 32 < 2048) fill(cur ^ 1, mm + 32);        // overlap with compute

    const u16* Kc = &Kbuf[cur][0];
    const u16* Vc = &Vbuf[cur][0];

    // ---- A = Qn Kn^T tiles: stream to HBM (NT) + bf16 stage into Sbuf ----
#pragma unroll
    for (int j = 0; j < 2; ++j) {
      Frag fb;
      v8f d = {};
      const u16* kr = Kc + (j * 16 + nl) * 64;
      fb.q[0] = *(const uint4*)(kr + ko);
      fb.q[1] = *(const uint4*)(kr + 16 + ko);
      d = wmma_bf16(faq0.v, fb.v, d);
      fb.q[0] = *(const uint4*)(kr + 32 + ko);
      fb.q[1] = *(const uint4*)(kr + 48 + ko);
      d = wmma_bf16(faq1.v, fb.v, d);
#pragma unroll
      for (int r = 0; r < 8; ++r) {
        const int M = hi * 8 + r;
        __builtin_nontemporal_store(
            d[r], &aout[(size_t)(rowbase + M) * 2048 + mm + j * 16 + nl]);
        sb[M * 32 + j * 16 + nl] = f2bf(d[r]);
      }
    }
    asm volatile("s_wait_dscnt 0x0" ::: "memory");  // RAW on per-wave staging

    // ---- A-tile as bf16 A-operand (16 rows x 32 m), from LDS ----
    Frag fa2;
    fa2.q[0] = *(const uint4*)&sb[nl * 32 + ko];
    fa2.q[1] = *(const uint4*)&sb[nl * 32 + 16 + ko];

    // ---- V B-operands from Vbuf; 4 output d-subtiles ----
    Frag fv;
    fv.q[0] = *(const uint4*)(Vc + (0  + nl) * 32 + ko);
    fv.q[1] = *(const uint4*)(Vc + (0  + nl) * 32 + ko + 16);
    o0 = wmma_bf16(fa2.v, fv.v, o0);
    fv.q[0] = *(const uint4*)(Vc + (16 + nl) * 32 + ko);
    fv.q[1] = *(const uint4*)(Vc + (16 + nl) * 32 + ko + 16);
    o1 = wmma_bf16(fa2.v, fv.v, o1);
    fv.q[0] = *(const uint4*)(Vc + (32 + nl) * 32 + ko);
    fv.q[1] = *(const uint4*)(Vc + (32 + nl) * 32 + ko + 16);
    o2 = wmma_bf16(fa2.v, fv.v, o2);
    fv.q[0] = *(const uint4*)(Vc + (48 + nl) * 32 + ko);
    fv.q[1] = *(const uint4*)(Vc + (48 + nl) * 32 + ko + 16);
    o3 = wmma_bf16(fa2.v, fv.v, o3);
  }

  // context out: [B,N,512] with feature offset h*64 (head merge done here)
  const int b = bh >> 3, h = bh & 7;
#pragma unroll
  for (int r = 0; r < 8; ++r) {
    float* Y = Yctx + ((size_t)b * 2048 + rowbase + hi * 8 + r) * 512 + h * 64;
    Y[0  + nl] = o0[r];
    Y[16 + nl] = o1[r];
    Y[32 + nl] = o2[r];
    Y[48 + nl] = o3[r];
  }
}

// ---------------------------------------------------------------------------
extern "C" void kernel_launch(void* const* d_in, const int* in_sizes, int n_in,
                              void* d_out, int out_size, void* d_ws,
                              size_t ws_size, hipStream_t stream) {
  const float* q  = (const float*)d_in[0];
  const float* k  = (const float*)d_in[1];
  const float* v  = (const float*)d_in[2];
  const float* Wq = (const float*)d_in[3];
  const float* bq = (const float*)d_in[4];
  const float* Wk = (const float*)d_in[5];
  const float* bk = (const float*)d_in[6];
  const float* Wv = (const float*)d_in[7];
  const float* bv = (const float*)d_in[8];
  const float* Wo = (const float*)d_in[9];
  const float* bo = (const float*)d_in[10];

  char* ws    = (char*)d_ws;
  float* Qf   = (float*)(ws);
  float* Kf   = (float*)(ws + (size_t)(8u  << 20));
  u16*   Vt   = (u16*)  (ws + (size_t)(16u << 20));
  u16*   Qn   = (u16*)  (ws + (size_t)(20u << 20));
  u16*   Kn   = (u16*)  (ws + (size_t)(24u << 20));
  float* Yctx = (float*)(ws + (size_t)(28u << 20));
  float* Ksum = (float*)(ws + (size_t)(36u << 20));

  const dim3 gg(32, 8), gb(256);
  gemm_proj<0><<<gg, gb, 0, stream>>>(q, Wq, bq, (void*)Qf);
  gemm_proj<1><<<gg, gb, 0, stream>>>(k, Wk, bk, (void*)Kf);
  gemm_proj<2><<<gg, gb, 0, stream>>>(v, Wv, bv, (void*)Vt);

  rownorm_q<<<4096, 256, 0, stream>>>(Qf, Qn);
  colsum_k<<<16, 256, 0, stream>>>(Kf, Ksum);
  norm_k<<<4096, 256, 0, stream>>>(Kf, Ksum, Kn);

  float* attnOut = (float*)d_out + (size_t)2 * 2048 * 512;  // after `out`
  attn_kernel<<<dim3(32, 16), 128, 0, stream>>>(Qn, Kn, Vt, attnOut, Yctx);

  gemm_proj<3><<<gg, gb, 0, stream>>>(Yctx, Wo, bo, d_out);
}
